// TMDLO_42683384988007
// MI455X (gfx1250) — matrix operand order repaired and checked by reading, compiled, not run on hardware
//
#include <hip/hip_runtime.h>
#include <hip/hip_bf16.h>

typedef _Float16 v16h __attribute__((ext_vector_type(16)));
typedef float    v8f  __attribute__((ext_vector_type(8)));
typedef float    v4f  __attribute__((ext_vector_type(4)));

#define VIEWS   6
#define BATCH   16384
#define DIM     512
#define CLASSES 100
#define CPAD    112   // 7 WMMA N-tiles of 16
#define LOSS_BLOCKS (BATCH / 8)   // 2048 blocks, 8 waves (samples) each

static constexpr size_t WH_ELEMS = (size_t)VIEWS * CPAD * DIM;            // 344064 f16
static constexpr size_t EV_OFF   = WH_ELEMS * 2;                          // 688128 B
static constexpr size_t EV_BYTES = (size_t)VIEWS * BATCH * CLASSES * 4;   // 39321600 B
static constexpr size_t PART_OFF = EV_OFF + EV_BYTES;                     // partials (2048 f32)

// ------------------------------------------- W f32 -> f16, padded to 112 rows
__global__ void TMDLO_convW(const float* __restrict__ W, _Float16* __restrict__ Wh) {
    size_t i = (size_t)blockIdx.x * blockDim.x + threadIdx.x;
    if (i >= WH_ELEMS) return;
    int    k = (int)(i % DIM);
    size_t t = i / DIM;
    int    c = (int)(t % CPAD);
    int    v = (int)(t / CPAD);
    float val = (c < CLASSES) ? W[((size_t)v * CLASSES + c) * DIM + k] : 0.0f;
    Wh[i] = (_Float16)val;
}

// ---------------------------------------- GEMM (f16 WMMA, fp32 acc) + softplus
// grid.x = VIEWS * (BATCH/128); 8 waves/block, each wave: 16 rows x 112 cols
__global__ __launch_bounds__(256) void TMDLO_gemm_softplus(
    const float*    __restrict__ X,     // [V][B][D] f32
    const _Float16* __restrict__ Wh,    // [V][CPAD][D] f16
    const float*    __restrict__ bias,  // [V][C] f32
    float*          __restrict__ ev)    // [V][B][C] f32
{
    const int wave   = threadIdx.x >> 5;
    const int lane   = threadIdx.x & 31;
    const int laneLo = lane & 15;
    const int laneHi = lane >> 4;

    const int v  = blockIdx.x / (BATCH / 128);
    const int rb = blockIdx.x % (BATCH / 128);
    const int r0 = rb * 128 + wave * 16;

    const float*    Xrow = X  + ((size_t)v * BATCH + r0 + laneLo) * DIM;
    const _Float16* Wv   = Wh + (size_t)v * CPAD * DIM;

    v8f acc[7];
#pragma unroll
    for (int n = 0; n < 7; ++n)
#pragma unroll
        for (int i = 0; i < 8; ++i) acc[n][i] = 0.0f;

    for (int k0 = 0; k0 < DIM; k0 += 32) {
        // --- A fragment (16x32 f16): lane holds M=laneLo,
        //     halves 0..7  = K (k0 + 8*laneHi)      .. +7
        //     halves 8..15 = K (k0 + 8*laneHi + 16) .. +7
        const float* pa = Xrow + k0 + laneHi * 8;
        v4f x0 = *(const v4f*)(pa + 0);
        v4f x1 = *(const v4f*)(pa + 4);
        v4f x2 = *(const v4f*)(pa + 16);
        v4f x3 = *(const v4f*)(pa + 20);
        v16h a;
#pragma unroll
        for (int i = 0; i < 4; ++i) {
            a[i]      = (_Float16)x0[i];
            a[4 + i]  = (_Float16)x1[i];
            a[8 + i]  = (_Float16)x2[i];
            a[12 + i] = (_Float16)x3[i];
        }
#pragma unroll
        for (int n = 0; n < 7; ++n) {
            // --- B fragment (32x16 f16): lane = column (class row of W),
            //     halves 0..7 = K k0+8*laneHi..+7, halves 8..15 = K k0+8*laneHi+16..+7
            const _Float16* pb = Wv + ((size_t)(n * 16 + laneLo)) * DIM + k0 + laneHi * 8;
            v16h bf;
#pragma unroll
            for (int i = 0; i < 8; ++i) { bf[i] = pb[i]; bf[8 + i] = pb[16 + i]; }
            acc[n] = __builtin_amdgcn_wmma_f32_16x16x32_f16(
                false, a, false, bf, (short)0, acc[n], false, false);
        }
    }

    // --- epilogue: bias + softplus, store only c < 100
#pragma unroll
    for (int n = 0; n < 7; ++n) {
        int c = n * 16 + laneLo;
        if (c < CLASSES) {
            float bb = bias[v * CLASSES + c];
#pragma unroll
            for (int i = 0; i < 8; ++i) {
                int   row = r0 + i + laneHi * 8;          // D layout: VGPR i -> M
                float x   = acc[n][i] + bb;
                float sp  = (x > 20.0f) ? x : log1pf(__expf(x));
                ev[((size_t)v * BATCH + row) * CLASSES + c] = sp;
            }
        }
    }
}

// ------------------------------------------------------------- loss reduction
__device__ __forceinline__ float TMDLO_wred(float x) {
#pragma unroll
    for (int o = 16; o > 0; o >>= 1) x += __shfl_xor(x, o, 32);
    return x;
}

// wave32 per sample; lane owns classes {lane, lane+32, lane+64, lane+96}.
// No atomics: per-block partial written to fixed slot (deterministic).
__global__ __launch_bounds__(256) void TMDLO_loss(
    const float* __restrict__ ev, const int* __restrict__ y,
    float* __restrict__ partials)
{
    const int wave = threadIdx.x >> 5;
    const int lane = threadIdx.x & 31;
    const int s    = blockIdx.x * 8 + wave;

    float e[VIEWS][4];
#pragma unroll
    for (int v = 0; v < VIEWS; ++v)
#pragma unroll
        for (int j = 0; j < 4; ++j) {
            int c = lane + 32 * j;
            e[v][j] = (c < CLASSES) ? ev[((size_t)v * BATCH + s) * CLASSES + c] : 0.0f;
        }
    const int label = y[s];

    // ---- accuracy loss
    float alpha[4], spart = 0.0f;
#pragma unroll
    for (int j = 0; j < 4; ++j) {
        int c = lane + 32 * j;
        float av = 1.0f;
#pragma unroll
        for (int v = 0; v < VIEWS; ++v) av += e[v][j];
        alpha[j] = av;
        if (c < CLASSES) spart += av;
    }
    const float S = TMDLO_wred(spart);
    float lap = 0.0f;
#pragma unroll
    for (int j = 0; j < 4; ++j) {
        int c = lane + 32 * j;
        if (c < CLASSES) {
            float P   = alpha[j] / S;
            float lab = (c == label) ? 1.0f : 0.0f;
            float d   = lab - P;
            lap += d * d + P * (1.0f - P) / (S + 1.0f);
        }
    }
    const float Lacc = TMDLO_wred(lap);

    // ---- consistency loss
    float Pv[VIEWS][4], H[VIEWS];
#pragma unroll
    for (int v = 0; v < VIEWS; ++v) {
        float svp = 0.0f;
#pragma unroll
        for (int j = 0; j < 4; ++j) {
            int c = lane + 32 * j;
            if (c < CLASSES) svp += e[v][j] + 1.0f;
        }
        float Sv = TMDLO_wred(svp);
        float hp = 0.0f;
#pragma unroll
        for (int j = 0; j < 4; ++j) {
            int c = lane + 32 * j;
            if (c < CLASSES) {
                float p = (e[v][j] + 1.0f) / Sv;
                Pv[v][j] = p;
                hp -= p * __log2f(p);
            } else {
                Pv[v][j] = 0.0f;
            }
        }
        H[v] = TMDLO_wred(hp);
    }

    float Lcon = 0.0f;
#pragma unroll
    for (int m = 0; m < VIEWS; ++m)
#pragma unroll
        for (int v = m + 1; v < VIEWS; ++v) {
            float hpp = 0.0f;
#pragma unroll
            for (int j = 0; j < 4; ++j) {
                int c = lane + 32 * j;
                if (c < CLASSES) {
                    float pp = 0.5f * (Pv[m][j] + Pv[v][j]);
                    hpp -= pp * __log2f(pp);
                }
            }
            float Hp = TMDLO_wred(hpp);
            Lcon += 2.0f * (Hp - 0.5f * (H[m] + H[v]));  // ordered pairs, mask diag
        }
    Lcon *= 1.0f / (float)(VIEWS - 1);

    __shared__ float wsum[8];
    if (lane == 0) wsum[wave] = Lacc + Lcon;
    __syncthreads();
    if (threadIdx.x == 0) {
        float t = 0.0f;
#pragma unroll
        for (int w = 0; w < 8; ++w) t += wsum[w];
        partials[blockIdx.x] = t;   // fixed slot per block -> deterministic
    }
}

// ---------------------- final deterministic reduction of 2048 block partials
__global__ __launch_bounds__(256) void TMDLO_final(
    const float* __restrict__ partials, float* __restrict__ out)
{
    const int wave = threadIdx.x >> 5;
    const int lane = threadIdx.x & 31;
    float t = 0.0f;
#pragma unroll
    for (int i = 0; i < LOSS_BLOCKS / 256; ++i)      // 8 strided, fixed order
        t += partials[threadIdx.x + i * 256];
    t = TMDLO_wred(t);
    __shared__ float wsum[8];
    if (lane == 0) wsum[wave] = t;
    __syncthreads();
    if (threadIdx.x == 0) {
        float r = 0.0f;
#pragma unroll
        for (int w = 0; w < 8; ++w) r += wsum[w];
        out[0] = r;
    }
}

extern "C" void kernel_launch(void* const* d_in, const int* in_sizes, int n_in,
                              void* d_out, int out_size, void* d_ws, size_t ws_size,
                              hipStream_t stream) {
    const float* X   = (const float*)d_in[0];
    const int*   y   = (const int*)d_in[1];
    const float* W   = (const float*)d_in[2];
    const float* b   = (const float*)d_in[3];
    float*       out = (float*)d_out;

    _Float16* Wh       = (_Float16*)d_ws;
    float*    ev       = (float*)((char*)d_ws + EV_OFF);
    float*    partials = (float*)((char*)d_ws + PART_OFF);

    TMDLO_convW<<<(int)((WH_ELEMS + 255) / 256), 256, 0, stream>>>(W, Wh);
    TMDLO_gemm_softplus<<<VIEWS * (BATCH / 128), 256, 0, stream>>>(X, Wh, b, ev);
    TMDLO_loss<<<LOSS_BLOCKS, 256, 0, stream>>>(ev, y, partials);
    TMDLO_final<<<1, 256, 0, stream>>>(partials, out);
}
